// NFM_81140522156065
// MI455X (gfx1250) — matrix operand order repaired and checked by reading, not compiled, not run on hardware
//
#include <hip/hip_runtime.h>
#include <math.h>

// ---------------- types for WMMA ----------------
typedef __attribute__((ext_vector_type(16))) __bf16 v16bf;
typedef __attribute__((ext_vector_type(8)))  float  v8f;

union BfOp { unsigned u[8]; v16bf v; };

#define BATCH 131072
#define FEAS 256
#define KDIM 256
#define ROWS_PER_BLOCK 128
#define D2 85
#define D2P 96
#define D3 64

// ---------------- LDS layout (bytes) ----------------
// Region A (reused): V^T during FM GEMM, tower weights afterwards.
#define VBT_OFF   0            // 256 x 264 bf16  = 135168
#define W1T_OFF   0            // 128 x 264 bf16  =  67584
#define W2T_OFF   67584        //  96 x 136 bf16  =  26112
#define W3T_OFF   93696        //  64 x 104 bf16  =  13312
#define H0_OFF    135168       // 128 x 264 bf16  =  67584
#define H1_OFF    202752       // 128 x 136 bf16  =  34816
#define H2_OFF    237568       // 128 x 104 bf16  =  26624
#define H3_OFF    264192       // 128 x  72 bf16  =  18432
#define LIN_OFF   282624       // 128 f32         =    512
#define SMEM_BYTES 283136      // 276.5 KB < 320 KB/WGP

// ---------------- bf16 helpers ----------------
__device__ __forceinline__ unsigned bf1(float f) {        // f32 -> bf16 (RNE)
  unsigned u = __float_as_uint(f);
  return (u + 0x7FFFu + ((u >> 16) & 1u)) >> 16;
}
__device__ __forceinline__ unsigned bfpack(float lo, float hi) {
  return bf1(lo) | (bf1(hi) << 16);
}
__device__ __forceinline__ float bfu(unsigned short h) { return __uint_as_float(((unsigned)h) << 16); }

// Packed bf16 square: one V_PK_MUL_BF16 (VOP3P op 42) instead of ~10 VALU ops.
__device__ __forceinline__ unsigned pk_sq_bf16(unsigned w) {
  unsigned r;
  asm("v_pk_mul_bf16 %0, %1, %1" : "=v"(r) : "v"(w));
  return r;
}

// One dense tower layer: out = relu(A @ W + b), all operands bf16 in LDS,
// wave handles rows [mbase, mbase+16). W stored transposed: W[n][k].
__device__ __forceinline__ void tower_layer(
    char* smem, int aOff, int aStride, int wOff, int wStride,
    int outOff, int outStride, const float* __restrict__ bias,
    int nReal, int ntiles, int ktiles, int mbase, int lrow, bool hi) {
  const unsigned short* A = (const unsigned short*)(smem + aOff);
  const unsigned short* W = (const unsigned short*)(smem + wOff);
  unsigned short* O = (unsigned short*)(smem + outOff);
  const unsigned short* arow = A + (mbase + lrow) * aStride;
  for (int nt = 0; nt < ntiles; ++nt) {
    v8f c = {0.f,0.f,0.f,0.f,0.f,0.f,0.f,0.f};
    const int N = nt * 16 + lrow;
    const unsigned short* brow = W + N * wStride;
    for (int kt = 0; kt < ktiles; ++kt) {
      // A layout (16-bit A 16x32): lanes 0-15 K={0..7,16..23}, lanes 16-31 K={8..15,24..31}
      const int k0 = kt * 32 + (hi ? 8 : 0);
      uint4 qa0 = *(const uint4*)(arow + k0);
      uint4 qa1 = *(const uint4*)(arow + k0 + 16);
      // B layout (16-bit B 32x16): lanes 0-15 K={0..7,16..23}, lanes 16-31 K={8..15,24..31}, N=lane%16
      const int f0 = kt * 32 + (hi ? 8 : 0);
      uint4 qb0 = *(const uint4*)(brow + f0);
      uint4 qb1 = *(const uint4*)(brow + f0 + 16);
      BfOp a, b;
      a.u[0]=qa0.x; a.u[1]=qa0.y; a.u[2]=qa0.z; a.u[3]=qa0.w;
      a.u[4]=qa1.x; a.u[5]=qa1.y; a.u[6]=qa1.z; a.u[7]=qa1.w;
      b.u[0]=qb0.x; b.u[1]=qb0.y; b.u[2]=qb0.z; b.u[3]=qb0.w;
      b.u[4]=qb1.x; b.u[5]=qb1.y; b.u[6]=qb1.z; b.u[7]=qb1.w;
      c = __builtin_amdgcn_wmma_f32_16x16x32_bf16(false, a.v, false, b.v,
                                                  (short)0, c, false, false);
    }
    const float bv = (N < nReal) ? bias[N] : 0.f;
#pragma unroll
    for (int j = 0; j < 8; ++j) {
      float h = c[j] + bv;
      h = h > 0.f ? h : 0.f;
      const int row = mbase + j + (hi ? 8 : 0);
      O[row * outStride + N] = (unsigned short)bf1(h);
    }
  }
}

__global__ void __launch_bounds__(256, 1)
nfm_fused(const float* __restrict__ x, const float* __restrict__ w_wide,
          const float* __restrict__ b_wide, const float* __restrict__ V,
          const float* __restrict__ w1, const float* __restrict__ b1,
          const float* __restrict__ w2, const float* __restrict__ b2,
          const float* __restrict__ w3, const float* __restrict__ b3,
          const float* __restrict__ w_out, const float* __restrict__ b_out,
          float* __restrict__ out) {
  extern __shared__ char smem[];
  const int tid  = threadIdx.x;
  const int lane = tid & 31;
  const int wave = tid >> 5;
  const int lrow = lane & 15;
  const bool hi  = lane >= 16;
  const int mbase = wave * 16;
  const long gbase = (long)blockIdx.x * ROWS_PER_BLOCK;

  // ---- Phase 0a: cooperative V^T -> bf16 LDS (padded stride 264) ----
  {
    unsigned short* vbt = (unsigned short*)(smem + VBT_OFF);
#pragma unroll 4
    for (int e = tid; e < FEAS * KDIM; e += 256) {
      const int f = e >> 8, n = e & 255;     // V[f][n], row-major global read
      vbt[n * 264 + f] = (unsigned short)bf1(V[e]);
    }
  }

  // ---- Phase 0b: wave loads its 16 rows of x as A-operands (bf16 + squared),
  //      plus the wide-linear partial dot in fp32. ----
  unsigned xa[8][8], x2a[8][8];
  float linp = 0.f;
  {
    const float* xr = x + (gbase + mbase + lrow) * (long)FEAS;
#pragma unroll
    for (int kt = 0; kt < 8; ++kt) {
      const int k0 = kt * 32 + (hi ? 8 : 0);
#pragma unroll
      for (int c = 0; c < 2; ++c) {
        const int base = k0 + c * 16;
        float4 f0 = *(const float4*)(xr + base);
        float4 f1 = *(const float4*)(xr + base + 4);
        xa[kt][c*4+0]  = bfpack(f0.x, f0.y);
        xa[kt][c*4+1]  = bfpack(f0.z, f0.w);
        xa[kt][c*4+2]  = bfpack(f1.x, f1.y);
        xa[kt][c*4+3]  = bfpack(f1.z, f1.w);
        x2a[kt][c*4+0] = bfpack(f0.x*f0.x, f0.y*f0.y);
        x2a[kt][c*4+1] = bfpack(f0.z*f0.z, f0.w*f0.w);
        x2a[kt][c*4+2] = bfpack(f1.x*f1.x, f1.y*f1.y);
        x2a[kt][c*4+3] = bfpack(f1.z*f1.z, f1.w*f1.w);
        linp += f0.x*w_wide[base+0] + f0.y*w_wide[base+1]
              + f0.z*w_wide[base+2] + f0.w*w_wide[base+3]
              + f1.x*w_wide[base+4] + f1.y*w_wide[base+5]
              + f1.z*w_wide[base+6] + f1.w*w_wide[base+7];
      }
    }
  }
  linp += __shfl_xor(linp, 16, 32);          // lanes L and L^16 cover disjoint K
  if (!hi) ((float*)(smem + LIN_OFF))[mbase + lrow] = linp + b_wide[0];

  __syncthreads();

  // ---- Phase 1: FM bi-interaction.  h0 = 0.5*((xV)^2 + (x^2)(V^2)) ----
  {
    const unsigned short* vbt = (const unsigned short*)(smem + VBT_OFF);
    unsigned short* h0 = (unsigned short*)(smem + H0_OFF);
    for (int nt = 0; nt < 16; ++nt) {
      v8f c1 = {0.f,0.f,0.f,0.f,0.f,0.f,0.f,0.f};
      v8f c2 = {0.f,0.f,0.f,0.f,0.f,0.f,0.f,0.f};
      const int N = nt * 16 + lrow;
      const unsigned short* brow = vbt + N * 264;
#pragma unroll
      for (int kt = 0; kt < 8; ++kt) {
        const int f0 = kt * 32 + (hi ? 8 : 0);
        uint4 q0 = *(const uint4*)(brow + f0);
        uint4 q1 = *(const uint4*)(brow + f0 + 16);
        BfOp a, a2, b, b2;
        b.u[0]=q0.x; b.u[1]=q0.y; b.u[2]=q0.z; b.u[3]=q0.w;
        b.u[4]=q1.x; b.u[5]=q1.y; b.u[6]=q1.z; b.u[7]=q1.w;
#pragma unroll
        for (int w = 0; w < 8; ++w) {
          b2.u[w] = pk_sq_bf16(b.u[w]);       // V^2 on the fly: 1x v_pk_mul_bf16
          a.u[w]  = xa[kt][w];
          a2.u[w] = x2a[kt][w];
        }
        c1 = __builtin_amdgcn_wmma_f32_16x16x32_bf16(false, a.v,  false, b.v,
                                                     (short)0, c1, false, false);
        c2 = __builtin_amdgcn_wmma_f32_16x16x32_bf16(false, a2.v, false, b2.v,
                                                     (short)0, c2, false, false);
      }
#pragma unroll
      for (int j = 0; j < 8; ++j) {
        const float h = 0.5f * (c1[j]*c1[j] + c2[j]);
        const int row = mbase + j + (hi ? 8 : 0);
        h0[row * 264 + N] = (unsigned short)bf1(h);
      }
    }
  }
  __syncthreads();   // region A (V^T) now dead

  // ---- Phase 2a: cooperative weight staging (transposed, zero-padded) ----
  {
    unsigned short* w1t = (unsigned short*)(smem + W1T_OFF);
    for (int e = tid; e < 256 * 128; e += 256) {         // w1[k][n]
      const int k = e >> 7, n = e & 127;
      w1t[n * 264 + k] = (unsigned short)bf1(w1[e]);
    }
    unsigned short* w2t = (unsigned short*)(smem + W2T_OFF);
    for (int e = tid; e < D2P * 128; e += 256) {         // w2[k][n], pad n 85..95
      const int n = e >> 7, k = e & 127;
      const float v = (n < D2) ? w2[k * D2 + n] : 0.f;
      w2t[n * 136 + k] = (unsigned short)bf1(v);
    }
    unsigned short* w3t = (unsigned short*)(smem + W3T_OFF);
    for (int e = tid; e < D3 * D2P; e += 256) {          // w3[k][n], pad k 85..95
      const int n = e / D2P, k = e - n * D2P;
      const float v = (k < D2) ? w3[k * D3 + n] : 0.f;
      w3t[n * 104 + k] = (unsigned short)bf1(v);
    }
  }
  __syncthreads();

  // ---- Phase 2b: deep tower (per-wave rows -> no barriers needed) ----
  tower_layer(smem, H0_OFF, 264, W1T_OFF, 264, H1_OFF, 136, b1, 128, 8, 8, mbase, lrow, hi);
  tower_layer(smem, H1_OFF, 136, W2T_OFF, 136, H2_OFF, 104, b2, D2, 6, 4, mbase, lrow, hi);
  tower_layer(smem, H2_OFF, 104, W3T_OFF, 104, H3_OFF,  72, b3, D3, 4, 3, mbase, lrow, hi);

  __syncthreads();

  // ---- Final: out = sigmoid(linear + h3 @ w_out + b_out) ----
  if (tid < ROWS_PER_BLOCK) {
    const unsigned short* h3 = (const unsigned short*)(smem + H3_OFF) + tid * 72;
    float s = 0.f;
#pragma unroll
    for (int n = 0; n < 64; ++n) s += bfu(h3[n]) * w_out[n];
    const float lin = ((const float*)(smem + LIN_OFF))[tid];
    const float z = lin + s + b_out[0];
    out[gbase + tid] = 1.f / (1.f + __expf(-z));
  }
}

extern "C" void kernel_launch(void* const* d_in, const int* in_sizes, int n_in,
                              void* d_out, int out_size, void* d_ws, size_t ws_size,
                              hipStream_t stream) {
  const float* x      = (const float*)d_in[0];
  const float* w_wide = (const float*)d_in[1];
  const float* b_wide = (const float*)d_in[2];
  const float* V      = (const float*)d_in[3];
  const float* w1     = (const float*)d_in[4];
  const float* b1     = (const float*)d_in[5];
  const float* w2     = (const float*)d_in[6];
  const float* b2     = (const float*)d_in[7];
  const float* w3     = (const float*)d_in[8];
  const float* b3     = (const float*)d_in[9];
  const float* w_out  = (const float*)d_in[10];
  const float* b_out  = (const float*)d_in[11];

  (void)hipFuncSetAttribute((const void*)nfm_fused,
                            hipFuncAttributeMaxDynamicSharedMemorySize, SMEM_BYTES);

  dim3 grid(BATCH / ROWS_PER_BLOCK);
  nfm_fused<<<grid, 256, SMEM_BYTES, stream>>>(x, w_wide, b_wide, V,
                                               w1, b1, w2, b2, w3, b3,
                                               w_out, b_out, (float*)d_out);
}
